// GQA_87402584473691
// MI455X (gfx1250) — matrix operand order repaired and checked by reading, not compiled, not run on hardware
//
#include <hip/hip_runtime.h>
#include <hip/hip_bf16.h>

// ---------------------------------------------------------------------------
// GQA attention for gfx1250 (MI455X): all matmuls via v_wmma_f32_16x16x32_f16,
// attention K-tiles staged to LDS by the Tensor Data Mover (tensor_load_to_lds
// + s_wait_tensorcnt), fp32 inputs converted to f16 on the fly.
// ---------------------------------------------------------------------------

#define EMBED  2048
#define SLEN   2048
#define BATCH  2
#define NHEADS 16
#define NKV    4
#define DHEAD  128
#define KVDIM  (DHEAD * NKV)   // 512

typedef __attribute__((ext_vector_type(16))) _Float16 v16h;
typedef __attribute__((ext_vector_type(8)))  float    v8f;
typedef __attribute__((ext_vector_type(4)))  unsigned v4u;
typedef __attribute__((ext_vector_type(8)))  int      v8i;
typedef __attribute__((ext_vector_type(4)))  int      v4i;

__device__ __forceinline__ v8f wmma_f16(v16h a, v16h b, v8f c) {
  // (neg_a, A, neg_b, B, c_mod, C, reuse_a, reuse_b)
  return __builtin_amdgcn_wmma_f32_16x16x32_f16(false, a, false, b, (short)0, c,
                                                false, false);
}

__device__ __forceinline__ v8f zero8() {
  v8f z;
#pragma unroll
  for (int i = 0; i < 8; ++i) z[i] = 0.0f;
  return z;
}

// A-fragment: 16x32 f16 (M x K), source row-major base[m*ld + k].
// Contiguous k order (this code, not the ISA table): lane m=lane&15, half=lane>>4; VGPR v holds K = 16*half + 2*v + {0,1}
__device__ __forceinline__ v16h load_frag_a(const _Float16* base, int ld, int lane) {
  const int m = lane & 15, half = lane >> 4;
  v16h a;
  unsigned* au = reinterpret_cast<unsigned*>(&a);
  const _Float16* p = base + (size_t)m * ld;
#pragma unroll
  for (int v = 0; v < 8; ++v) {
    const int k = (half << 4) + (v << 1);
    au[v] = *reinterpret_cast<const unsigned*>(p + k);
  }
  return a;
}

// B-fragment: 32x16 f16 (K x N), source stored N-major: base[n*ld + k].
// Contiguous k order (as load_frag_a): lane n=lane&15; lanes 0-15 hold K=0..15, lanes 16-31 hold K=16..31.
__device__ __forceinline__ v16h load_frag_b(const _Float16* base, int ld, int lane) {
  const int n = lane & 15, half = lane >> 4;
  v16h b;
  unsigned* bu = reinterpret_cast<unsigned*>(&b);
  const _Float16* p = base + (size_t)n * ld + (half << 4);
#pragma unroll
  for (int j = 0; j < 8; ++j)
    bu[j] = *reinterpret_cast<const unsigned*>(p + (j << 1));
  return b;
}

// ---------------------------------------------------------------------------
// Tensor Data Mover: 2-D f16 tile (tile_d0 x tile_d1 elements) -> LDS, with
// optional LDS row padding.  D# packed per CDNA5 ISA.
// Must be executed by a single wave; completion via s_wait_tensorcnt.
// This toolchain's builtin is the 6-arg form:
//   (uint32x4 g0, int32x8 g1, int32x4 g2, int32x4 g3, int32x8, i32 cpol)
// ---------------------------------------------------------------------------
__device__ __forceinline__ void tdm_load_2d_f16(
    unsigned lds_byte_addr, const void* gaddr, unsigned tensor_d0,
    unsigned tensor_d1, unsigned tile_d0, unsigned tile_d1, unsigned stride_d0,
    bool pad_en, unsigned pad_interval, unsigned pad_amount) {
  const unsigned long long ga = (unsigned long long)gaddr;
  v4u g0;
  g0[0] = 1u;                                            // count=1 (valid, user)
  g0[1] = lds_byte_addr;                                 // lds_addr
  g0[2] = (unsigned)(ga & 0xFFFFFFFFu);                  // global_addr[31:0]
  g0[3] = (unsigned)((ga >> 32) & 0x01FFFFFFu) | (2u << 30);  // [56:32] | type=2
  v8i g1;
  g1[0] = (int)((1u << 16) |                             // data_size = 2 bytes
                (pad_en ? (1u << 20) : 0u) |             // pad_enable
                ((pad_interval & 7u) << 22) |            // 8<<pi bytes / pad
                ((pad_amount & 127u) << 25));            // (pa+1) dwords pad
  g1[1] = (int)((tensor_d0 & 0xFFFFu) << 16);            // tensor_dim0[15:0]
  g1[2] = (int)(((tensor_d0 >> 16) & 0xFFFFu) |          // tensor_dim0[31:16]
                ((tensor_d1 & 0xFFFFu) << 16));          // tensor_dim1[15:0]
  g1[3] = (int)(((tensor_d1 >> 16) & 0xFFFFu) |          // tensor_dim1[31:16]
                ((tile_d0 & 0xFFFFu) << 16));            // tile_dim0
  g1[4] = (int)(tile_d1 & 0xFFFFu);                      // tile_dim1 (dim2=0)
  g1[5] = (int)stride_d0;                                // tensor_dim0_stride lo
  g1[6] = 0;                                             // stride0 hi | stride1 lo
  g1[7] = 0;
  const v4i z4 = {0, 0, 0, 0};                           // 2-D: groups 2/3 unused
  const v8i z8 = {0, 0, 0, 0, 0, 0, 0, 0};
  __builtin_amdgcn_tensor_load_to_lds(g0, g1, z4, z4, z8, 0);
}

// ---------------------------------------------------------------------------
// Tiled GEMM: C[M,N] = scale * (A[M,K] @ B[K,N]);  A is fp32 or f16, B fp32.
// Block tile 128x64, K-chunk 32, 8 waves; each wave owns a 32x32 tile
// (2 A-frags x 2 B-frags -> 4 WMMA per chunk).
// HEADMAJOR: write f16 to [b][h][s][d] (h = n/Dsz, s = m%Ssz) for attention.
// ---------------------------------------------------------------------------
template <typename TA, bool HEADMAJOR>
__global__ __launch_bounds__(256) void gemm_wmma_kernel(
    const TA* __restrict__ A, const float* __restrict__ Bw,
    void* __restrict__ Cout, int M, int N, int K, int Ssz, int Dsz,
    float scale) {
  __shared__ _Float16 lA[128][34];  // [m][k]
  __shared__ _Float16 lB[64][34];   // [n][k]  (transposed -> contiguous K)

  const int tid  = threadIdx.x;
  const int lane = tid & 31, w = tid >> 5;
  const int lm = lane & 15, lh = lane >> 4;
  const int m0 = blockIdx.y * 128;
  const int n0 = blockIdx.x * 64;
  const int tr = (w & 3) * 32;      // wave row base within block tile
  const int tc = (w >> 2) * 32;     // wave col base within block tile

  v8f acc[2][2];
#pragma unroll
  for (int i = 0; i < 2; ++i)
#pragma unroll
    for (int j = 0; j < 2; ++j) acc[i][j] = zero8();

  for (int kc = 0; kc < K; kc += 32) {
    __syncthreads();
    {  // stage A tile (128x32) as f16: thread -> one half-row of 16
      const int row = tid >> 1, kk0 = (tid & 1) << 4;
      const TA* src = A + (size_t)(m0 + row) * K + kc + kk0;
#pragma unroll
      for (int j = 0; j < 16; ++j)
        lA[row][kk0 + j] = (_Float16)(float)src[j];
    }
    {  // stage B tile (32x64) transposed as f16
      const int k = tid >> 3, nn0 = (tid & 7) << 3;
      const float* src = Bw + (size_t)(kc + k) * N + n0 + nn0;
#pragma unroll
      for (int j = 0; j < 8; ++j)
        lB[nn0 + j][k] = (_Float16)src[j];
    }
    __syncthreads();

    const v16h a0 = load_frag_a(&lA[tr][0], 34, lane);
    const v16h a1 = load_frag_a(&lA[tr + 16][0], 34, lane);
    const v16h b0 = load_frag_b(&lB[tc][0], 34, lane);
    const v16h b1 = load_frag_b(&lB[tc + 16][0], 34, lane);
    acc[0][0] = wmma_f16(a0, b0, acc[0][0]);
    acc[0][1] = wmma_f16(a0, b1, acc[0][1]);
    acc[1][0] = wmma_f16(a1, b0, acc[1][0]);
    acc[1][1] = wmma_f16(a1, b1, acc[1][1]);
  }

  const int Hn = HEADMAJOR ? (N / Dsz) : 0;
#pragma unroll
  for (int i = 0; i < 2; ++i)
#pragma unroll
    for (int j = 0; j < 2; ++j) {
#pragma unroll
      for (int r = 0; r < 8; ++r) {
        const int m = m0 + tr + i * 16 + r + lh * 8;
        const int n = n0 + tc + j * 16 + lm;
        const float val = acc[i][j][r] * scale;
        if constexpr (HEADMAJOR) {
          _Float16* out = (_Float16*)Cout;
          const int bb = m / Ssz, ss = m % Ssz;
          out[(((size_t)bb * Hn + n / Dsz) * Ssz + ss) * Dsz + n % Dsz] =
              (_Float16)val;
        } else {
          ((float*)Cout)[(size_t)m * N + n] = val;
        }
      }
    }
}

// ---------------------------------------------------------------------------
// Flash attention: one block = (b,h) x 128 query rows; 8 waves x 16-row tiles.
// K tile staged by TDM (with LDS pad -> 130-half rows), V staged transposed.
// Dynamic LDS so the TDM descriptor's lds_addr offsets start at 0.
//   [0,        16640)  lK  [64][130]
//   [16640,    33536)  lVt [128][66]
//   [33536,    49920)  lP  [8][16][64]
// ---------------------------------------------------------------------------
#define LDS_K_OFF  0
#define LDS_VT_OFF 16640
#define LDS_P_OFF  33536
#define LDS_BYTES  49920

__global__ __launch_bounds__(256) void attn_wmma_kernel(
    const _Float16* __restrict__ qf,   // [B][H][S][D]   (pre-scaled)
    const _Float16* __restrict__ kf,   // [B][Hkv][S][D]
    const _Float16* __restrict__ vf,   // [B][Hkv][S][D]
    const float* __restrict__ mask,    // [B][S]
    _Float16* __restrict__ attn) {     // [B][S][H*D]
  extern __shared__ _Float16 smem[];
  _Float16(*lK)[130] = reinterpret_cast<_Float16(*)[130]>(smem);
  _Float16(*lVt)[66] =
      reinterpret_cast<_Float16(*)[66]>((char*)smem + LDS_VT_OFF);
  _Float16(*lP)[16][64] =
      reinterpret_cast<_Float16(*)[16][64]>((char*)smem + LDS_P_OFF);

  const int bh = blockIdx.x;
  const int b = bh / NHEADS, h = bh % NHEADS;
  const int hkv = h / (NHEADS / NKV);
  const int s0 = blockIdx.y * 128;
  const int tid = threadIdx.x, w = tid >> 5, lane = tid & 31;
  const int lm = lane & 15, lh = lane >> 4;

  const _Float16* qh = qf + ((size_t)b * NHEADS + h) * SLEN * DHEAD;
  const _Float16* kh = kf + ((size_t)b * NKV + hkv) * SLEN * DHEAD;
  const _Float16* vh = vf + ((size_t)b * NKV + hkv) * SLEN * DHEAD;

  // Preload this wave's Q fragments straight from global (row-major A source)
  const int row0 = s0 + w * 16;
  v16h qa[4];
#pragma unroll
  for (int c = 0; c < 4; ++c)
    qa[c] = load_frag_a(qh + (size_t)row0 * DHEAD + c * 32, DHEAD, lane);

  v8f accO[8];
#pragma unroll
  for (int dt = 0; dt < 8; ++dt) accO[dt] = zero8();
  float mrow[8], lrow[8];
#pragma unroll
  for (int r = 0; r < 8; ++r) { mrow[r] = -3.0e38f; lrow[r] = 0.0f; }

  for (int kb = 0; kb < SLEN; kb += 64) {
    __syncthreads();
    // ---- K block: Tensor Data Mover, 64 rows x 128 f16, pad 1 dword per
    //      64-dword row -> LDS rows of 130 halves (bank-conflict-free B-frags)
    if (w == 0) {
      tdm_load_2d_f16(LDS_K_OFF, kh + (size_t)kb * DHEAD,
                      /*tensor_d0=*/DHEAD, /*tensor_d1=*/SLEN - kb,
                      /*tile_d0=*/DHEAD, /*tile_d1=*/64, /*stride_d0=*/DHEAD,
                      /*pad_en=*/true, /*pad_interval=*/5, /*pad_amount=*/0);
      __builtin_amdgcn_s_wait_tensorcnt(0);
    }
    // ---- V block staged transposed: lVt[d][key]
    for (int i = tid; i < 64 * 128; i += 256) {
      const int key = i >> 7, d = i & 127;
      lVt[d][key] = vh[(size_t)(kb + key) * DHEAD + d];
    }
    if (kb + 64 < SLEN)  // pull the next K block toward the caches
      __builtin_prefetch(kh + (size_t)(kb + 64) * DHEAD + (tid << 3), 0, 1);
    __syncthreads();

    // ---- scores: 4 tiles of 16x16 over this wave's 16 rows
    v8f sc[4];
#pragma unroll
    for (int t = 0; t < 4; ++t) {
      v8f c = zero8();
#pragma unroll
      for (int kc = 0; kc < 4; ++kc) {
        const v16h bk = load_frag_b(&lK[t * 16][kc * 32], 130, lane);
        c = wmma_f16(qa[kc], bk, c);
      }
      const float madd =
          (1.0f - mask[(size_t)b * SLEN + kb + t * 16 + lm]) * -1e9f;
#pragma unroll
      for (int r = 0; r < 8; ++r) c[r] += madd;
      sc[t] = c;
    }

    // ---- online softmax (rows live across 16 lanes of a half-wave)
    float mnew[8], alpha[8], rsum[8];
#pragma unroll
    for (int r = 0; r < 8; ++r) {
      float mx = fmaxf(fmaxf(sc[0][r], sc[1][r]), fmaxf(sc[2][r], sc[3][r]));
#pragma unroll
      for (int off = 1; off <= 8; off <<= 1)
        mx = fmaxf(mx, __shfl_xor(mx, off, 32));
      mnew[r] = fmaxf(mrow[r], mx);
      alpha[r] = __expf(mrow[r] - mnew[r]);
      rsum[r] = 0.0f;
    }
#pragma unroll
    for (int t = 0; t < 4; ++t)
#pragma unroll
      for (int r = 0; r < 8; ++r) {
        const float p = __expf(sc[t][r] - mnew[r]);
        rsum[r] += p;
        lP[w][r + lh * 8][t * 16 + lm] = (_Float16)p;  // C-layout -> A-layout
      }
#pragma unroll
    for (int r = 0; r < 8; ++r) {
#pragma unroll
      for (int off = 1; off <= 8; off <<= 1)
        rsum[r] += __shfl_xor(rsum[r], off, 32);
      lrow[r] = lrow[r] * alpha[r] + rsum[r];
      mrow[r] = mnew[r];
#pragma unroll
      for (int dt = 0; dt < 8; ++dt) accO[dt][r] *= alpha[r];
    }

    // ---- O += P @ V   (per-wave LDS, DS ops in-order within a wave)
#pragma unroll
    for (int kc = 0; kc < 2; ++kc) {
      const v16h pa = load_frag_a(&lP[w][0][kc * 32], 64, lane);
#pragma unroll
      for (int dt = 0; dt < 8; ++dt) {
        const v16h bv = load_frag_b(&lVt[dt * 16][kc * 32], 66, lane);
        accO[dt] = wmma_f16(pa, bv, accO[dt]);
      }
    }
  }

  // ---- normalize and write f16, laid out [b][s][h*D+d] for the O-projection
#pragma unroll
  for (int r = 0; r < 8; ++r) {
    const float inv = 1.0f / lrow[r];
    const int srow = row0 + r + lh * 8;
#pragma unroll
    for (int dt = 0; dt < 8; ++dt)
      attn[((size_t)b * SLEN + srow) * EMBED + h * DHEAD + dt * 16 + lm] =
          (_Float16)(accO[dt][r] * inv);
  }
}

// ---------------------------------------------------------------------------
extern "C" void kernel_launch(void* const* d_in, const int* in_sizes, int n_in,
                              void* d_out, int out_size, void* d_ws,
                              size_t ws_size, hipStream_t stream) {
  const float* query = (const float*)d_in[0];
  const float* key   = (const float*)d_in[1];
  const float* value = (const float*)d_in[2];
  const float* mask  = (const float*)d_in[3];
  const float* Wq    = (const float*)d_in[4];
  const float* Wk    = (const float*)d_in[5];
  const float* Wv    = (const float*)d_in[6];
  const float* Wo    = (const float*)d_in[7];
  float* out = (float*)d_out;

  char* ws = (char*)d_ws;
  _Float16* qf = (_Float16*)(ws);                            // 16 MB [B][H][S][D]
  _Float16* kf = (_Float16*)(ws + (size_t)16 * 1024 * 1024); //  4 MB [B][Hkv][S][D]
  _Float16* vf = (_Float16*)(ws + (size_t)20 * 1024 * 1024); //  4 MB
  _Float16* at = (_Float16*)(ws + (size_t)24 * 1024 * 1024); // 16 MB [B*S][E]

  const int M = BATCH * SLEN;  // 4096
  const float qscale = 0.08838834764831845f;  // 1/sqrt(128) folded into Q
  dim3 blk(256);

  // projections (fp32 in -> f16 head-major out)
  gemm_wmma_kernel<float, true><<<dim3(EMBED / 64, M / 128), blk, 0, stream>>>(
      query, Wq, qf, M, EMBED, EMBED, SLEN, DHEAD, qscale);
  gemm_wmma_kernel<float, true><<<dim3(KVDIM / 64, M / 128), blk, 0, stream>>>(
      key, Wk, kf, M, KVDIM, EMBED, SLEN, DHEAD, 1.0f);
  gemm_wmma_kernel<float, true><<<dim3(KVDIM / 64, M / 128), blk, 0, stream>>>(
      value, Wv, vf, M, KVDIM, EMBED, SLEN, DHEAD, 1.0f);

  // flash attention (dynamic LDS so TDM lds_addr offsets start at 0)
  attn_wmma_kernel<<<dim3(BATCH * NHEADS, SLEN / 128), blk, LDS_BYTES, stream>>>(
      qf, kf, vf, mask, at);

  // output projection (f16 A, fp32 out)
  gemm_wmma_kernel<_Float16, false><<<dim3(EMBED / 64, M / 128), blk, 0, stream>>>(
      at, Wo, out, M, EMBED, EMBED, 0, 1, 1.0f);
}
